// TwoSimplicialAttention_47193100649132
// MI455X (gfx1250) — matrix operand order repaired and checked by reading, not compiled, not run on hardware
//
#include <hip/hip_runtime.h>

typedef __attribute__((ext_vector_type(16))) __bf16 v16bf;
typedef __attribute__((ext_vector_type(8)))  float  v8f;

#define N_    256
#define D_    64
#define QS    68      // f32 stride for Q rows (conflict-free banks)
#define K1S   72      // bf16 stride for K1 rows: 144B rows -> 16B aligned, conflict-free
#define V1S   264     // bf16 stride for V1^T rows: 528B rows -> 16B aligned, conflict-free

// LDS layout (bytes), all 16B aligned
#define OFF_Q    0           // float [16][QS]        = 4352
#define OFF_K1   4352        // bf16  [256][K1S]      = 36864
#define OFF_V1T  41216       // bf16  [80][V1S]       = 42240 (rows 64..79: ones/zeros l-tile)
#define OFF_M    83456       // float [4][16]         = 256
#define OFF_L    83712       // float [4][16]         = 256
#define OFF_O    83968       // float [4][16][64]     = 16384
#define SMEM_BYTES 100352

#define LOG2E 1.44269504088896340736f

__device__ __forceinline__ __bf16 f2bf(float f) {       // staging only (cold)
  union { float f; unsigned u; } x; x.f = f;
  unsigned r = (x.u + 0x7FFFu + ((x.u >> 16) & 1u)) >> 16;
  unsigned short s = (unsigned short)r;
  __bf16 o; __builtin_memcpy(&o, &s, 2); return o;
}

// swap bits 3<->4 of a j index: the in-chunk permutation that makes the
// accumulator's (hf*8+r) j-split line up with a contiguous (hf*16+e) B split; unused: bp is in table order.
__device__ __forceinline__ int swap34(int x) {
  return (x & ~24) | ((x & 8) << 1) | ((x & 16) >> 1);
}

// pack two f32 -> dword of two bf16, round-to-nearest-ish (1 perm + 2 adds)
__device__ __forceinline__ unsigned pack_bf16_rne(float lo, float hi) {
  unsigned a, b;
  __builtin_memcpy(&a, &lo, 4);
  __builtin_memcpy(&b, &hi, 4);
  return __builtin_amdgcn_perm(b + 0x8000u, a + 0x8000u, 0x07060302u);
}
// truncating pack (1 op) -- used for P (non-negative, bias cancels in num/denom)
__device__ __forceinline__ unsigned pack_bf16_trunc(float lo, float hi) {
  unsigned a, b;
  __builtin_memcpy(&a, &lo, 4);
  __builtin_memcpy(&b, &hi, 4);
  return __builtin_amdgcn_perm(b, a, 0x07060302u);
}

__device__ __forceinline__ v16bf frag_from(const unsigned* u) {
  union { unsigned u[8]; v16bf v; } t;
#pragma unroll
  for (int i = 0; i < 8; ++i) t.u[i] = u[i];
  return t.v;
}

// A-operand fragment (16x32 bf16): lane half hf -> elements p[0..7], p[16..23]
__device__ __forceinline__ v16bf load_a_frag(const __bf16* p) {
  uint4 c0 = *(const uint4*)(p);
  uint4 c1 = *(const uint4*)(p + 16);
  unsigned u[8] = {c0.x, c0.y, c0.z, c0.w, c1.x, c1.y, c1.z, c1.w};
  return frag_from(u);
}

__device__ __forceinline__ v8f wmma_bf16(v16bf a, v16bf b, v8f c) {
  return __builtin_amdgcn_wmma_f32_16x16x32_bf16(false, a, false, b, (short)0, c,
                                                 false, false);
}

// launch_bounds(128, 2): two waves per SIMD (<=512 VGPRs). The per-iteration
// chunk rotation below makes operand addresses kt-dependent so the allocator
// can neither hoist the 256-VGPR K1 fragment set nor spill it to scratch
// (the round-4 failure mode): operands stay as in-loop ds_load_b128.
__global__ void __launch_bounds__(128, 2)
two_simplicial_attn(const float* __restrict__ qg, const float* __restrict__ k1g,
                    const float* __restrict__ v1g, const float* __restrict__ k2g,
                    const float* __restrict__ v2g, float* __restrict__ outg) {
  extern __shared__ char smem[];
  float*  sQ   = (float*)(smem + OFF_Q);
  __bf16* sK1  = (__bf16*)(smem + OFF_K1);
  __bf16* sV1t = (__bf16*)(smem + OFF_V1T);
  float*  sM   = (float*)(smem + OFF_M);
  float*  sL   = (float*)(smem + OFF_L);
  float*  sO   = (float*)(smem + OFF_O);

  const int tid  = threadIdx.x;            // 128 threads = 4 waves (wave32)
  const int bh   = blockIdx.x >> 4;
  const int qt   = blockIdx.x & 15;
  const int q0   = qt * 16;
  const int wave = tid >> 5;
  const int lane = tid & 31;
  const int hf   = lane >> 4;
  const int ln   = lane & 15;
  const int kb   = hf * 8;                 // A-operand K-base for this half

  const size_t base = (size_t)bh * N_ * D_;

  // ---- stage: Q*log2e (f32), K1 (bf16), V1^T + l-tile (bf16) -------
  for (int i = tid; i < 16 * D_; i += 128) {
    int m = i >> 6, d = i & 63;
    sQ[m * QS + d] = qg[base + (size_t)(q0 + m) * D_ + d] * LOG2E;
  }
  for (int i = tid; i < N_ * D_; i += 128) {
    int j = i >> 6, d = i & 63;
    sK1[j * K1S + d] = f2bf(k1g[base + (size_t)j * D_ + d]);
    sV1t[d * V1S + j] = f2bf(v1g[base + (size_t)j * D_ + d]);
  }
  for (int i = tid; i < 16 * V1S; i += 128) {     // rows 64..79: ones row + zeros
    int r = i / V1S, c = i % V1S;
    sV1t[(64 + r) * V1S + c] = f2bf(r == 0 ? 1.0f : 0.0f);
  }
  __syncthreads();

  // per-lane softmax state: this lane owns query q = ln (log2 domain)
  float m_s = -1e30f, l_s = 0.f;
  float oacc[4][8];                        // out^T: d = dt*16 + hf*8 + r, q = ln
#pragma unroll
  for (int dt = 0; dt < 4; ++dt)
#pragma unroll
    for (int r = 0; r < 8; ++r) oacc[dt][r] = 0.f;

  for (int kt = 0; kt < 64; ++kt) {
    const int kk  = wave * 64 + kt;        // each wave owns a 64-slice of k2
    const int rot = kt & 7;                // per-iter j-chunk rotation (uniform)
    const float* k2row = k2g + base + (size_t)kk * D_;
    const float* v2row = v2g + base + (size_t)kk * D_;
    if (kt + 1 < 64) {
      __builtin_prefetch(k2row + D_, 0, 0);   // global_prefetch_b8
      __builtin_prefetch(v2row + D_, 0, 0);
    }

    // ---- B = (Q ⊙ k2[kk])^T: lane -> column q=ln, K=d = F*32 + hf*8 + e (e<8), +16 (e>=8) ----
    v16bf bq[2];
#pragma unroll
    for (int F = 0; F < 2; ++F) {
      const int d0 = F * 32 + hf * 8;
      const float4* qv = (const float4*)(sQ + ln * QS + d0);
      const float4* kv = (const float4*)(k2row + d0);
      unsigned pkq[8];
#pragma unroll
      for (int i = 0; i < 4; ++i) {
        float4 a = qv[i + (i & 2)], b = kv[i + (i & 2)];
        pkq[2 * i]     = pack_bf16_rne(a.x * b.x, a.y * b.y);
        pkq[2 * i + 1] = pack_bf16_rne(a.z * b.z, a.w * b.w);
      }
      bq[F] = frag_from(pkq);
    }

    // ---- S^T: A = K1 (M=j), B = bq (N=q), K = d = 64 ------------------------
    // position ct processes physical chunk pc = (ct+rot)&7 (order-invariant)
    v8f S[16];
#pragma unroll
    for (int ct = 0; ct < 8; ++ct) {
      const int pc = (ct + rot) & 7;
      const __bf16* crow = sK1 + (pc * 32 + ln) * K1S;   // tile 2*pc row base
#pragma unroll
      for (int t = 0; t < 2; ++t) {
        const __bf16* row = crow + t * (16 * K1S);
        v16bf a0 = load_a_frag(row + kb);
        v16bf a1 = load_a_frag(row + 32 + kb);
        v8f c = {};
        c = wmma_bf16(a0, bq[0], c);
        c = wmma_bf16(a1, bq[1], c);
        S[2 * ct + t] = c;                 // j = pc*32 + t*16 + hf*8 + r
      }
    }

    // ---- online max (per-lane over j; one cross-half shuffle) ---------------
    float mx = m_s;
#pragma unroll
    for (int jt = 0; jt < 16; ++jt)
#pragma unroll
      for (int r = 0; r < 8; ++r) mx = fmaxf(mx, S[jt][r]);
    mx = fmaxf(mx, __shfl_xor(mx, 16, 32));
    const float corr = __builtin_amdgcn_exp2f(m_s - mx);
    m_s = mx;

    // ---- P = 2^(S-m), packed bf16 per tile (truncating pack) ----------------
    unsigned pk[16][4];
#pragma unroll
    for (int jt = 0; jt < 16; ++jt) {
      float p[8];
#pragma unroll
      for (int r = 0; r < 8; ++r) p[r] = __builtin_amdgcn_exp2f(S[jt][r] - mx);
#pragma unroll
      for (int i = 0; i < 4; ++i)
        pk[jt][i] = pack_bf16_trunc(p[2 * i], p[2 * i + 1]);
    }
#pragma unroll
    for (int dt = 0; dt < 4; ++dt)
#pragma unroll
      for (int r = 0; r < 8; ++r) oacc[dt][r] *= corr;

    // ---- PV: out^T = V1^T (A, M=d, incl. ones-row l-tile) @ P^T (B, N=q) ----
    v8f tacc[5];
#pragma unroll
    for (int dt = 0; dt < 5; ++dt) tacc[dt] = (v8f){};
#pragma unroll
    for (int jc = 0; jc < 8; ++jc) {       // position jc -> physical chunk pc
      const int pc = (jc + rot) & 7;       // matches the S-GEMM rotation
      unsigned bfr[8];
#pragma unroll
      for (int i = 0; i < 4; ++i) {
        bfr[i]     = pk[2 * jc][i];
        bfr[4 + i] = pk[2 * jc + 1][i];
      }
      v16bf bp = frag_from(bfr);
#pragma unroll
      for (int dt = 0; dt < 5; ++dt) {     // dt==4 is the Σ_j P (denominator) tile
        const __bf16* vrow = sV1t + (dt * 16 + ln) * V1S + pc * 32;
        tacc[dt] = wmma_bf16(load_a_frag(vrow + kb), bp, tacc[dt]);
      }
    }
    // l update from the ones-row tile: only (hf=0, r=0) holds the sum
    float lsum = tacc[4][0];
    lsum += __shfl_xor(lsum, 16, 32);
    l_s = l_s * corr + lsum;
    // scale by v2[kk, d] and accumulate
#pragma unroll
    for (int dt = 0; dt < 4; ++dt) {
      const float4* vv = (const float4*)(v2row + dt * 16 + hf * 8);
      float4 v0 = vv[0], v1 = vv[1];
      oacc[dt][0] += tacc[dt][0] * v0.x;  oacc[dt][1] += tacc[dt][1] * v0.y;
      oacc[dt][2] += tacc[dt][2] * v0.z;  oacc[dt][3] += tacc[dt][3] * v0.w;
      oacc[dt][4] += tacc[dt][4] * v1.x;  oacc[dt][5] += tacc[dt][5] * v1.y;
      oacc[dt][6] += tacc[dt][6] * v1.z;  oacc[dt][7] += tacc[dt][7] * v1.w;
    }
  }

  // ---- merge the 4 per-wave partial softmax states --------------------------
  if (hf == 0) {                           // stats replicated across halves
    sM[wave * 16 + ln] = m_s;
    sL[wave * 16 + ln] = l_s;
  }
#pragma unroll
  for (int dt = 0; dt < 4; ++dt)
#pragma unroll
    for (int r = 0; r < 8; ++r)
      sO[(wave * 16 + ln) * D_ + dt * 16 + hf * 8 + r] = oacc[dt][r];
  __syncthreads();

  for (int i = tid; i < 16 * D_; i += 128) {
    int m = i >> 6, d = i & 63;
    float gm = -1e30f;
#pragma unroll
    for (int w = 0; w < 4; ++w) gm = fmaxf(gm, sM[w * 16 + m]);
    float L = 0.f, val = 0.f;
#pragma unroll
    for (int w = 0; w < 4; ++w) {
      float e = __builtin_amdgcn_exp2f(sM[w * 16 + m] - gm);  // log2 domain
      L   += sL[w * 16 + m] * e;
      val += sO[(w * 16 + m) * D_ + d] * e;
    }
    outg[base + (size_t)(q0 + m) * D_ + d] = val / L;
  }
}

extern "C" void kernel_launch(void* const* d_in, const int* in_sizes, int n_in,
                              void* d_out, int out_size, void* d_ws, size_t ws_size,
                              hipStream_t stream) {
  (void)in_sizes; (void)n_in; (void)out_size; (void)d_ws; (void)ws_size;
  const float* q  = (const float*)d_in[0];
  const float* k1 = (const float*)d_in[1];
  const float* v1 = (const float*)d_in[2];
  const float* k2 = (const float*)d_in[3];
  const float* v2 = (const float*)d_in[4];
  float* out = (float*)d_out;

  (void)hipFuncSetAttribute((const void*)two_simplicial_attn,
                            hipFuncAttributeMaxDynamicSharedMemorySize,
                            SMEM_BYTES);
  // grid: (B*H)=16  x  (N/16)=16 query tiles -> 256 workgroups, 4 waves each
  two_simplicial_attn<<<dim3(256), dim3(128), SMEM_BYTES, stream>>>(
      q, k1, v1, k2, v2, out);
}
